// SnEncoder_5059471475016
// MI455X (gfx1250) — matrix operand order repaired and checked by reading, not compiled, not run on hardware
//
#include <hip/hip_runtime.h>
#include <hip/hip_bf16.h>

// ---------------------------------------------------------------------------
// SnEncoder fused implementation for MI455X (gfx1250, wave32, WMMA).
//
// Shapes: B=16, n=32, Cin=16, Cn=8, H=64, Cout=64.
// out1[b,i,j,k,d] = Eij[b,i,j,d]+Eik[b,i,k,d]+Ejk[b,j,k,d]+Fi[b,i,d]+Fj[b,j,d]
//                   +Fk[b,k,d]+G[b,d]  (derived analytically from W1 slices).
// h1 = relu(bn1(out1)) is recomputed on the fly everywhere (never stored:
// saves 2x134MB of traffic). Layer 2's only dense contraction h1 @ W2[0:64]
// uses v_wmma_f32_16x16x32_f16 with f16 operands staged in LDS; the weight
// matrix is kept TRANSPOSED in LDS so B-fragments are contiguous 32-byte
// per-lane loads (ds_load_b128 instead of 16x ds_load_u16).
// ---------------------------------------------------------------------------

typedef __attribute__((ext_vector_type(16))) _Float16 v16h;
typedef __attribute__((ext_vector_type(8)))  float    v8f;

#define DEVI __device__ __forceinline__

constexpr int   Bc   = 16, Nn = 32, Hc = 64, Cin = 16, Cn = 8;
constexpr float INV_N   = 1.0f / 32.0f;
constexpr float INV_N2  = 1.0f / 1024.0f;
constexpr float INV_N3  = 1.0f / 32768.0f;
constexpr float INV_BN3 = 1.0f / 524288.0f;   // B*n^3
constexpr float EPSf    = 1e-5f;

constexpr size_t SZ_E = (size_t)Bc * Nn * Nn * Hc;  // 1,048,576 floats
constexpr size_t SZ_F = (size_t)Bc * Nn * Hc;       //    32,768 floats
constexpr size_t SZ_G = (size_t)Bc * Hc;            //     1,024 floats

// ------------------------- WMMA fragment helpers ---------------------------
// A-matrix 16x32 f16, contiguous k order (same as B): lane L holds row M=L%16;
//   lanes 0-15: elems 0..15 -> K k0+0..15
//   lanes 16-31: shift K base by +16.
DEVI v16h load_a_frag(const _Float16* Alds, int ldA, int k0, int lane) {
  int m  = lane & 15;
  int kb = k0 + ((lane >> 4) << 4);
  const _Float16* p = Alds + m * ldA + kb;
  v16h a;
#pragma unroll
  for (int e = 0; e < 8; ++e)  a[e] = p[e];        // K kb..kb+7   (16B vector)
#pragma unroll
  for (int e = 8; e < 16; ++e) a[e] = p[e];        // K kb+8..+15 (16B vector)
  return a;
}
// B-matrix 32x16 f16: lane L holds column N=n0+L%16;
//   lanes 0-15: K k0+e, lanes 16-31: K k0+16+e.
// Wt is stored TRANSPOSED in LDS: Wt[n*ldB + k] => contiguous per-lane load.
DEVI v16h load_b_frag_t(const _Float16* Wt, int ldB, int n0, int k0, int lane) {
  int nn = n0 + (lane & 15);
  int kb = k0 + ((lane >> 4) << 4);
  const _Float16* p = Wt + nn * ldB + kb;
  v16h b;
#pragma unroll
  for (int e = 0; e < 16; ++e) b[e] = p[e];        // 32 contiguous bytes
  return b;
}

// ------------------------- K1a: axis means of x ----------------------------
__global__ void k_means(const float* __restrict__ x, float* __restrict__ m1,
                        float* __restrict__ m0, float* __restrict__ mm) {
  int b = blockIdx.x, t = threadIdx.x;                 // block 256
  for (int idx = t; idx < Nn * Cin; idx += 256) {      // m1[b,i,c]=mean_j
    int i = idx >> 4, c = idx & 15;
    float s = 0.f;
    for (int j = 0; j < Nn; ++j) s += x[(((size_t)(b * Nn + i) * Nn) + j) * Cin + c];
    m1[(b * Nn + i) * Cin + c] = s * INV_N;
  }
  for (int idx = t; idx < Nn * Cin; idx += 256) {      // m0[b,j,c]=mean_i
    int j = idx >> 4, c = idx & 15;
    float s = 0.f;
    for (int i = 0; i < Nn; ++i) s += x[(((size_t)(b * Nn + i) * Nn) + j) * Cin + c];
    m0[(b * Nn + j) * Cin + c] = s * INV_N;
  }
  __syncthreads();
  if (t < Cin) {                                       // mm[b,c]
    float s = 0.f;
    for (int i = 0; i < Nn; ++i) s += m1[(b * Nn + i) * Cin + t];
    mm[b * Cin + t] = s * INV_N;
  }
}

// ------------- K1b: E tensors (x2 rows x combined W1 slices) ---------------
__global__ void k_e_tensors(const float* __restrict__ x, const float* __restrict__ W1,
                            float* __restrict__ Eij, float* __restrict__ Eik,
                            float* __restrict__ Ejk) {
  int row = blockIdx.x;                                // (b,p,q), block 64
  int d = threadIdx.x;
  __shared__ float xr[Cin];
  if (d < Cin) xr[d] = x[(size_t)row * Cin + d];
  __syncthreads();
  float eij = 0.f, eik = 0.f, ejk = 0.f;
#pragma unroll
  for (int c = 0; c < Cin; ++c) {
    float xv = xr[c];
    eij += xv * (W1[(c)       * Hc + d] + W1[(144 + c) * Hc + d]);
    eik += xv * (W1[(16 + c)  * Hc + d] + W1[(112 + c) * Hc + d]);
    ejk += xv * (W1[(32 + c)  * Hc + d] + W1[(80 + c)  * Hc + d]);
  }
  Eij[(size_t)row * Hc + d] = eij;
  Eik[(size_t)row * Hc + d] = eik;
  Ejk[(size_t)row * Hc + d] = ejk;
}

// ------------- K1c: F/G tensors (means & x_node x W1/Wn1 combos) -----------
__global__ void k_fg_tensors(const float* __restrict__ m1, const float* __restrict__ m0,
                             const float* __restrict__ mm, const float* __restrict__ xn,
                             const float* __restrict__ W1, const float* __restrict__ Wn1,
                             float* __restrict__ Fi, float* __restrict__ Fj,
                             float* __restrict__ Fk, float* __restrict__ G) {
  int row = blockIdx.x, d = threadIdx.x;               // (b,p), block 64
  int b = row >> 5, p = row & 31;
  float fi = 0.f, fj = 0.f, fk = 0.f;
#pragma unroll
  for (int c = 0; c < Cin; ++c) {
    float v1 = m1[row * Cin + c], v0 = m0[row * Cin + c];
    fi += v1 * (W1[(96 + c) * Hc + d] + W1[(160 + c) * Hc + d] +
                W1[(192 + c) * Hc + d] + W1[(208 + c) * Hc + d]);
    fj += v0 * (W1[(48 + c) * Hc + d] + W1[(240 + c) * Hc + d]) +
          v1 * (W1[(176 + c) * Hc + d] + W1[(272 + c) * Hc + d]);
    fk += v0 * (W1[(64 + c) * Hc + d] + W1[(128 + c) * Hc + d] +
                W1[(304 + c) * Hc + d] + W1[(320 + c) * Hc + d]);
  }
#pragma unroll
  for (int c = 0; c < Cn; ++c) {
    float xv = xn[row * Cn + c];
    fi += xv * Wn1[c * Hc + d];
    fj += xv * Wn1[(8 + c) * Hc + d];
    fk += xv * Wn1[(16 + c) * Hc + d];
  }
  Fi[row * Hc + d] = fi; Fj[row * Hc + d] = fj; Fk[row * Hc + d] = fk;
  if (p == 0) {
    float g = 0.f;
#pragma unroll
    for (int c = 0; c < Cin; ++c)
      g += mm[b * Cin + c] * (W1[(224 + c) * Hc + d] + W1[(256 + c) * Hc + d] +
                              W1[(288 + c) * Hc + d] + W1[(336 + c) * Hc + d] +
                              W1[(352 + c) * Hc + d] + W1[(368 + c) * Hc + d]);
    G[b * Hc + d] = g;
  }
}

// ---------------- K2: BN1 batch stats over virtual out1 --------------------
__global__ void k_bn1_stats(const float* __restrict__ Eij, const float* __restrict__ Eik,
                            const float* __restrict__ Ejk, const float* __restrict__ Fi,
                            const float* __restrict__ Fj, const float* __restrict__ Fk,
                            const float* __restrict__ G,
                            float* __restrict__ gsum, float* __restrict__ gsq) {
  int wg = blockIdx.x, t = threadIdx.x;                // (b,i,j), block 256
  int j = wg & 31, i = (wg >> 5) & 31, b = wg >> 10;
  __shared__ float cS[Hc], cQ[Hc];
  if (t < Hc) { cS[t] = 0.f; cQ[t] = 0.f; }
  __syncthreads();
  int d = t & 63;
  float fixed = Eij[((size_t)wg) * Hc + d] + Fi[(b * Nn + i) * Hc + d] +
                Fj[(b * Nn + j) * Hc + d] + G[b * Hc + d];
  float s = 0.f, q = 0.f;
  for (int k = (t >> 6); k < Nn; k += 4) {
    float o = fixed + Eik[((size_t)(b * Nn + i) * Nn + k) * Hc + d] +
              Ejk[((size_t)(b * Nn + j) * Nn + k) * Hc + d] +
              Fk[(b * Nn + k) * Hc + d];
    s += o; q += o * o;
  }
  atomicAdd(&cS[d], s); atomicAdd(&cQ[d], q);
  __syncthreads();
  if (t < Hc) { atomicAdd(gsum + t, cS[t]); atomicAdd(gsq + t, cQ[t]); }
}

// ------------------- BN finalize: scale/shift from stats -------------------
__global__ void k_bn_finalize(const float* __restrict__ gsum, const float* __restrict__ gsq,
                              const float* __restrict__ g, const float* __restrict__ bta,
                              float invN, float* __restrict__ a, float* __restrict__ c) {
  int d = threadIdx.x;                                  // block 64
  float mu = gsum[d] * invN;
  float var = gsq[d] * invN - mu * mu;
  float s = g[d] * rsqrtf(var + EPSf);
  a[d] = s; c[d] = bta[d] - mu * s;
}

// -------- K4: axis sums of h1 (mode 0: sum_i, 1: sum_j, 2: sum_k) ----------
__global__ void k_h1_reduce(const float* __restrict__ Eij, const float* __restrict__ Eik,
                            const float* __restrict__ Ejk, const float* __restrict__ Fi,
                            const float* __restrict__ Fj, const float* __restrict__ Fk,
                            const float* __restrict__ G, const float* __restrict__ a1,
                            const float* __restrict__ c1, float* __restrict__ S, int mode) {
  int wg = blockIdx.x, d = threadIdx.x;                // (b,p,q), block 64
  int q = wg & 31, p = (wg >> 5) & 31, b = wg >> 10;
  float ad = a1[d], cd = c1[d], gd = G[b * Hc + d];
  float acc = 0.f;
  for (int u = 0; u < Nn; ++u) {
    int i, j, k;
    if (mode == 0)      { i = u; j = p; k = q; }
    else if (mode == 1) { i = p; j = u; k = q; }
    else                { i = p; j = q; k = u; }
    float o = Eij[((size_t)(b * Nn + i) * Nn + j) * Hc + d] +
              Eik[((size_t)(b * Nn + i) * Nn + k) * Hc + d] +
              Ejk[((size_t)(b * Nn + j) * Nn + k) * Hc + d] +
              Fi[(b * Nn + i) * Hc + d] + Fj[(b * Nn + j) * Hc + d] +
              Fk[(b * Nn + k) * Hc + d] + gd;
    acc += fmaxf(ad * o + cd, 0.f);
  }
  S[(size_t)wg * Hc + d] = acc;
}

// -------------- K5a: order-1 means of h1 from S1/S2 ------------------------
__global__ void k_r_prep(const float* __restrict__ S1, const float* __restrict__ S2,
                         float* __restrict__ R23, float* __restrict__ R13,
                         float* __restrict__ R12) {
  int row = blockIdx.x, d = threadIdx.x;               // (b,p), block 64
  int b = row >> 5, p = row & 31;
  float a23 = 0.f, a13 = 0.f, a12 = 0.f;
  for (int u = 0; u < Nn; ++u) {
    a23 += S2[((size_t)(b * Nn + p) * Nn + u) * Hc + d];  // sum_k of sum_j h1
    a13 += S1[((size_t)(b * Nn + p) * Nn + u) * Hc + d];  // sum_k of sum_i h1
    a12 += S1[((size_t)(b * Nn + u) * Nn + p) * Hc + d];  // sum_j of sum_i h1
  }
  R23[row * Hc + d] = a23 * INV_N2;
  R13[row * Hc + d] = a13 * INV_N2;
  R12[row * Hc + d] = a12 * INV_N2;
}
__global__ void k_r123(const float* __restrict__ R23, float* __restrict__ R123) {
  int b = blockIdx.x, d = threadIdx.x;                 // block 64
  float s = 0.f;
  for (int i = 0; i < Nn; ++i) s += R23[(b * Nn + i) * Hc + d];
  R123[b * Hc + d] = s * INV_N;
}

// -------- Generic WMMA GEMM: C[rows,64] = (A[rows,KK]*scale) @ W[KK,64] ----
template <int KK>
__global__ void k_gemm_wmma(const float* __restrict__ A, const float* __restrict__ W,
                            float* __restrict__ C, float scaleA) {
  // grid = rows/16, block = 128 (4 waves, one 16-col N-tile each).
  __shared__ _Float16 Alds[16 * KK];
  __shared__ _Float16 Wt[Hc * KK];       // TRANSPOSED: Wt[n*KK + k]
  int r0 = blockIdx.x * 16, t = threadIdx.x;
  __builtin_prefetch(W, 0, 1);                         // global_prefetch_b8
#pragma unroll 4
  for (int idx = t; idx < 16 * KK; idx += 128) {
    int r = idx / KK, k = idx % KK;                    // compile-time KK
    Alds[r * KK + k] = (_Float16)(A[(size_t)(r0 + r) * KK + k] * scaleA);
  }
#pragma unroll 4
  for (int idx = t; idx < KK * Hc; idx += 128) {
    int k = idx >> 6, n = idx & 63;
    Wt[n * KK + k] = (_Float16)W[idx];
  }
  __syncthreads();
  int lane = t & 31, n0 = (t >> 5) * 16;
  v8f acc = {};
#pragma unroll
  for (int k0 = 0; k0 < KK; k0 += 32) {
    v16h a = load_a_frag(Alds, KK, k0, lane);
    v16h b = load_b_frag_t(Wt, KK, n0, k0, lane);
    acc = __builtin_amdgcn_wmma_f32_16x16x32_f16(false, a, false, b,
                                                 (short)0, acc, false, false);
  }
  int col = n0 + (lane & 15), rb = (lane >> 4) * 8;
#pragma unroll
  for (int vr = 0; vr < 8; ++vr)
    C[(size_t)(r0 + rb + vr) * Hc + col] = acc[vr];
}

// ---- K6/K8: fused layer-2 WMMA pass over virtual out2 ---------------------
// mode 0: accumulate BN2 per-channel sum/sumsq.
// mode 1: h2 = relu(bn2(out2)); accumulate pair sums T23/T13/T12.
__global__ void k_layer2_pass(
    const float* __restrict__ Eij, const float* __restrict__ Eik,
    const float* __restrict__ Ejk, const float* __restrict__ Fi,
    const float* __restrict__ Fj, const float* __restrict__ Fk,
    const float* __restrict__ G, const float* __restrict__ a1,
    const float* __restrict__ c1, const float* __restrict__ V0,
    const float* __restrict__ E2ij, const float* __restrict__ E2ik,
    const float* __restrict__ E2jk, const float* __restrict__ F2i,
    const float* __restrict__ F2j, const float* __restrict__ F2k,
    const float* __restrict__ G2, const float* __restrict__ a2,
    const float* __restrict__ c2, float* __restrict__ gsum,
    float* __restrict__ gsq, float* __restrict__ T23, float* __restrict__ T13,
    float* __restrict__ T12, int mode) {
  // grid = B*n*n*2 = 32768; block = 128 (4 waves). WG handles (b,i,j), k-tile.
  int wg = blockIdx.x, t = threadIdx.x;
  int kh = wg & 1, j = (wg >> 1) & 31, i = (wg >> 6) & 31, b = wg >> 11;
  int k0g = kh * 16;
  __shared__ _Float16 Alds[16 * Hc];
  __shared__ _Float16 Wt[Hc * Hc];       // TRANSPOSED V0
  __shared__ float t12part[16 * Hc];
  __shared__ float red0[Hc], red1[Hc];
  if (t < Hc) { red0[t] = 0.f; red1[t] = 0.f; }
  for (int idx = t; idx < 16 * Hc; idx += 128) t12part[idx] = 0.f;

  // Stage h1 (recomputed from broadcast tensors) as f16 A-tile: rows = k.
  const float* eij = Eij + ((size_t)(b * Nn + i) * Nn + j) * Hc;
  const float* fi = Fi + (b * Nn + i) * Hc;
  const float* fj = Fj + (b * Nn + j) * Hc;
  const float* gb = G + b * Hc;
  for (int idx = t; idx < 16 * Hc; idx += 128) {
    int r = idx >> 6, c = idx & 63;
    int k = k0g + r;
    float o = eij[c] + Eik[((size_t)(b * Nn + i) * Nn + k) * Hc + c] +
              Ejk[((size_t)(b * Nn + j) * Nn + k) * Hc + c] +
              fi[c] + fj[c] + Fk[(b * Nn + k) * Hc + c] + gb[c];
    Alds[r * Hc + c] = (_Float16)fmaxf(a1[c] * o + c1[c], 0.f);
  }
  for (int idx = t; idx < Hc * Hc; idx += 128) {
    int k = idx >> 6, n = idx & 63;
    Wt[n * Hc + k] = (_Float16)V0[idx];
  }
  __syncthreads();

  int lane = t & 31, n0 = (t >> 5) * 16;
  v8f acc = {};
#pragma unroll
  for (int k0 = 0; k0 < Hc; k0 += 32) {
    v16h a = load_a_frag(Alds, Hc, k0, lane);
    v16h bfr = load_b_frag_t(Wt, Hc, n0, k0, lane);
    acc = __builtin_amdgcn_wmma_f32_16x16x32_f16(false, a, false, bfr,
                                                 (short)0, acc, false, false);
  }

  int d = n0 + (lane & 15), rb = (lane >> 4) * 8;
  float base = E2ij[((size_t)(b * Nn + i) * Nn + j) * Hc + d] +
               F2i[(b * Nn + i) * Hc + d] + F2j[(b * Nn + j) * Hc + d] +
               G2[b * Hc + d];
  if (mode == 0) {
    float s = 0.f, q = 0.f;
#pragma unroll
    for (int vr = 0; vr < 8; ++vr) {
      int k = k0g + rb + vr;
      float o2 = acc[vr] + base + E2ik[((size_t)(b * Nn + i) * Nn + k) * Hc + d] +
                 E2jk[((size_t)(b * Nn + j) * Nn + k) * Hc + d] +
                 F2k[(b * Nn + k) * Hc + d];
      s += o2; q += o2 * o2;
    }
    atomicAdd(&red0[d], s); atomicAdd(&red1[d], q);
    __syncthreads();
    if (t < Hc) { atomicAdd(gsum + t, red0[t]); atomicAdd(gsq + t, red1[t]); }
  } else {
    float s23 = 0.f;
#pragma unroll
    for (int vr = 0; vr < 8; ++vr) {
      int k = k0g + rb + vr;
      float o2 = acc[vr] + base + E2ik[((size_t)(b * Nn + i) * Nn + k) * Hc + d] +
                 E2jk[((size_t)(b * Nn + j) * Nn + k) * Hc + d] +
                 F2k[(b * Nn + k) * Hc + d];
      float h2 = fmaxf(a2[d] * o2 + c2[d], 0.f);
      s23 += h2;
      t12part[(rb + vr) * Hc + d] = h2;                // unique writer per slot
    }
    atomicAdd(&red0[d], s23);
    __syncthreads();
    if (t < Hc) {
      atomicAdd(T23 + (b * Nn + i) * Hc + t, red0[t]);
      atomicAdd(T13 + (b * Nn + j) * Hc + t, red0[t]);
    }
    for (int idx = t; idx < 16 * Hc; idx += 128) {
      int r = idx >> 6, c = idx & 63;
      atomicAdd(T12 + (b * Nn + (k0g + r)) * Hc + c, t12part[idx]);
    }
  }
}

// ------------- K8b: assemble [t23|t13|t12|t123] rows for final GEMM --------
__global__ void k_acat(const float* __restrict__ T23, const float* __restrict__ T13,
                       const float* __restrict__ T12, float* __restrict__ Acat) {
  int row = blockIdx.x, d = threadIdx.x;               // (b,m), block 64
  int b = row >> 5;
  float t123 = 0.f;                                    // sum_i of sum_{j,k} h2
  for (int ii = 0; ii < Nn; ++ii) t123 += T23[(b * Nn + ii) * Hc + d];
  Acat[(size_t)row * 256 + d]        = T23[row * Hc + d] * INV_N2;  // mean_{jk}
  Acat[(size_t)row * 256 + 64 + d]   = T13[row * Hc + d] * INV_N2;  // mean_{ik}
  Acat[(size_t)row * 256 + 128 + d]  = T12[row * Hc + d] * INV_N2;  // mean_{ij}
  Acat[(size_t)row * 256 + 192 + d]  = t123 * INV_N3;               // mean_{ijk}
}

// ------------------- K9: BN1d over [B*n, Cout] + output --------------------
__global__ void k_bn1d_final(const float* __restrict__ pre, const float* __restrict__ g,
                             const float* __restrict__ bta, float* __restrict__ out) {
  int d = blockIdx.x, t = threadIdx.x;                 // grid 64, block 256
  __shared__ float sS[256], sQ[256];
  float v1 = pre[(size_t)t * Hc + d];
  float v2 = pre[(size_t)(t + 256) * Hc + d];
  sS[t] = v1 + v2; sQ[t] = v1 * v1 + v2 * v2;
  __syncthreads();
  for (int off = 128; off > 0; off >>= 1) {
    if (t < off) { sS[t] += sS[t + off]; sQ[t] += sQ[t + off]; }
    __syncthreads();
  }
  float mu = sS[0] * (1.f / 512.f);
  float var = sQ[0] * (1.f / 512.f) - mu * mu;
  float a = g[d] * rsqrtf(var + EPSf);
  float c = bta[d] - mu * a;
  out[(size_t)t * Hc + d] = a * v1 + c;
  out[(size_t)(t + 256) * Hc + d] = a * v2 + c;
}

// ---------------------------------------------------------------------------
extern "C" void kernel_launch(void* const* d_in, const int* in_sizes, int n_in,
                              void* d_out, int out_size, void* d_ws, size_t ws_size,
                              hipStream_t stream) {
  const float* x   = (const float*)d_in[0];
  const float* xn  = (const float*)d_in[1];
  const float* W1  = (const float*)d_in[2];
  const float* Wn1 = (const float*)d_in[3];
  const float* g1  = (const float*)d_in[4];
  const float* b1  = (const float*)d_in[5];
  const float* W2  = (const float*)d_in[6];
  const float* g2  = (const float*)d_in[7];
  const float* b2  = (const float*)d_in[8];
  const float* Wc  = (const float*)d_in[9];
  const float* gc  = (const float*)d_in[10];
  const float* bc  = (const float*)d_in[11];
  float* out = (float*)d_out;
  float* ws  = (float*)d_ws;

  size_t o = 0;
  float* Eij = ws + o; o += SZ_E;  float* Eik = ws + o; o += SZ_E;
  float* Ejk = ws + o; o += SZ_E;
  float* Fi  = ws + o; o += SZ_F;  float* Fj  = ws + o; o += SZ_F;
  float* Fk  = ws + o; o += SZ_F;  float* G   = ws + o; o += SZ_G;
  float* m1  = ws + o; o += (size_t)Bc * Nn * Cin;
  float* m0  = ws + o; o += (size_t)Bc * Nn * Cin;
  float* mm  = ws + o; o += (size_t)Bc * Cin;
  float* bn1sum = ws + o; o += 64; float* bn1sq = ws + o; o += 64;
  float* a1v = ws + o; o += 64;    float* c1v = ws + o; o += 64;
  float* S1 = ws + o; o += SZ_E;   float* S2 = ws + o; o += SZ_E;
  float* S3 = ws + o; o += SZ_E;
  float* E2jk = ws + o; o += SZ_E; float* E2ik = ws + o; o += SZ_E;
  float* E2ij = ws + o; o += SZ_E;
  float* R23 = ws + o; o += SZ_F;  float* R13 = ws + o; o += SZ_F;
  float* R12 = ws + o; o += SZ_F;  float* R123 = ws + o; o += SZ_G;
  float* F2i = ws + o; o += SZ_F;  float* F2j = ws + o; o += SZ_F;
  float* F2k = ws + o; o += SZ_F;  float* G2  = ws + o; o += SZ_G;
  float* bn2sum = ws + o; o += 64; float* bn2sq = ws + o; o += 64;
  float* a2v = ws + o; o += 64;    float* c2v = ws + o; o += 64;
  float* T23 = ws + o; o += SZ_F;  float* T13 = ws + o; o += SZ_F;
  float* T12 = ws + o; o += SZ_F;
  float* Acat = ws + o; o += (size_t)512 * 256;
  float* pre  = ws + o; o += (size_t)512 * Hc;

  // Zero accumulation buffers (bn1sum..bn1sq, bn2sum..bn2sq, T23..T12 each
  // contiguous).
  hipMemsetAsync(bn1sum, 0, 2 * 64 * sizeof(float), stream);
  hipMemsetAsync(bn2sum, 0, 2 * 64 * sizeof(float), stream);
  hipMemsetAsync(T23, 0, 3 * SZ_F * sizeof(float), stream);

  // Layer-1 analytic broadcast tensors.
  k_means<<<Bc, 256, 0, stream>>>(x, m1, m0, mm);
  k_e_tensors<<<Bc * Nn * Nn, 64, 0, stream>>>(x, W1, Eij, Eik, Ejk);
  k_fg_tensors<<<Bc * Nn, 64, 0, stream>>>(m1, m0, mm, xn, W1, Wn1, Fi, Fj, Fk, G);

  // BN1.
  k_bn1_stats<<<Bc * Nn * Nn, 256, 0, stream>>>(Eij, Eik, Ejk, Fi, Fj, Fk, G,
                                                bn1sum, bn1sq);
  k_bn_finalize<<<1, 64, 0, stream>>>(bn1sum, bn1sq, g1, b1, INV_BN3, a1v, c1v);

  // h1 axis sums.
  k_h1_reduce<<<Bc * Nn * Nn, 64, 0, stream>>>(Eij, Eik, Ejk, Fi, Fj, Fk, G,
                                               a1v, c1v, S1, 0);
  k_h1_reduce<<<Bc * Nn * Nn, 64, 0, stream>>>(Eij, Eik, Ejk, Fi, Fj, Fk, G,
                                               a1v, c1v, S2, 1);
  k_h1_reduce<<<Bc * Nn * Nn, 64, 0, stream>>>(Eij, Eik, Ejk, Fi, Fj, Fk, G,
                                               a1v, c1v, S3, 2);
  k_r_prep<<<Bc * Nn, 64, 0, stream>>>(S1, S2, R23, R13, R12);
  k_r123<<<Bc, 64, 0, stream>>>(R23, R123);

  // Layer-2 broadcast tensors via WMMA GEMMs. V_p = W2 + p*64*64.
  k_gemm_wmma<64><<<1024, 128, 0, stream>>>(S1, W2 + 1 * 4096, E2jk, INV_N);
  k_gemm_wmma<64><<<1024, 128, 0, stream>>>(S2, W2 + 2 * 4096, E2ik, INV_N);
  k_gemm_wmma<64><<<1024, 128, 0, stream>>>(S3, W2 + 3 * 4096, E2ij, INV_N);
  k_gemm_wmma<64><<<32, 128, 0, stream>>>(R23, W2 + 4 * 4096, F2i, 1.f);
  k_gemm_wmma<64><<<32, 128, 0, stream>>>(R13, W2 + 5 * 4096, F2j, 1.f);
  k_gemm_wmma<64><<<32, 128, 0, stream>>>(R12, W2 + 6 * 4096, F2k, 1.f);
  k_gemm_wmma<64><<<1, 128, 0, stream>>>(R123, W2 + 7 * 4096, G2, 1.f);

  // Fused layer-2 passes (h1 recomputed on the fly, main GEMM via WMMA).
  k_layer2_pass<<<Bc * Nn * Nn * 2, 128, 0, stream>>>(
      Eij, Eik, Ejk, Fi, Fj, Fk, G, a1v, c1v, W2,
      E2ij, E2ik, E2jk, F2i, F2j, F2k, G2, a2v, c2v,
      bn2sum, bn2sq, T23, T13, T12, 0);
  k_bn_finalize<<<1, 64, 0, stream>>>(bn2sum, bn2sq, g2, b2, INV_BN3, a2v, c2v);
  k_layer2_pass<<<Bc * Nn * Nn * 2, 128, 0, stream>>>(
      Eij, Eik, Ejk, Fi, Fj, Fk, G, a1v, c1v, W2,
      E2ij, E2ik, E2jk, F2i, F2j, F2k, G2, a2v, c2v,
      bn2sum, bn2sq, T23, T13, T12, 1);

  // Layer-3 contraction + BN1d.
  k_acat<<<Bc * Nn, 64, 0, stream>>>(T23, T13, T12, Acat);
  k_gemm_wmma<256><<<32, 128, 0, stream>>>(Acat, Wc, pre, 1.f);
  k_bn1d_final<<<64, 256, 0, stream>>>(pre, gc, bc, out);
}
